// Transformer_37847251813102
// MI455X (gfx1250) — matrix operand order repaired and checked by reading, not compiled, not run on hardware
//
#include <hip/hip_runtime.h>
#include <hip/hip_bf16.h>

// ---------------- problem constants ----------------
#define D     256
#define H     8
#define DH    32
#define DFF   2048
#define NQ    36
#define QPAD  48
#define NE    8
#define WORD  20
#define SSUB  32768
#define STOT  32928         // 32768 + 20*8
#define NCHUNK 32           // shared-key chunks (1024 keys each)
#define CHUNK  1024

typedef __attribute__((ext_vector_type(16))) __bf16 v16bf;
typedef __attribute__((ext_vector_type(8)))  float  v8f;

union ABfrag { v16bf v; unsigned short s[16]; };

__device__ __forceinline__ unsigned short f2bf(float f) {
  union { float f; unsigned u; } v; v.f = f;
  unsigned r = v.u + 0x7fffu + ((v.u >> 16) & 1u);
  return (unsigned short)(r >> 16);
}
__device__ __forceinline__ float bf2f(unsigned short s) {
  union { unsigned u; float f; } v; v.u = ((unsigned)s) << 16; return v.f;
}
__device__ __forceinline__ v8f vzero8() {
  v8f z; 
#pragma unroll
  for (int i = 0; i < 8; ++i) z[i] = 0.f;
  return z;
}
__device__ __forceinline__ v8f wmma_bf16(v16bf a, v16bf b, v8f c) {
  return __builtin_amdgcn_wmma_f32_16x16x32_bf16(false, a, false, b, (short)0, c, false, false);
}
// A-fragment (16x32 bf16) builder, own k order (not the ISA table order):
// lane<16: K = k0+[0..15]; lane>=16: K = k0+[16..31] (contiguous, matches B).
__device__ __forceinline__ v16bf load_a16(const unsigned short* X, int stride,
                                          int row, int k0, int lhalf) {
  ABfrag a;
  int base = row * stride + k0 + 16 * lhalf;
#pragma unroll
  for (int j = 0; j < 8; ++j) a.s[j] = X[base + j];
#pragma unroll
  for (int j = 0; j < 8; ++j) a.s[8 + j] = X[base + 8 + j];
  return a.v;
}

// ---------------- kernel 1: weight f32 -> bf16 ----------------
__global__ void wconv_kernel(const float* __restrict__ ca_w_in,
                             const float* __restrict__ w1, const float* __restrict__ w2,
                             unsigned short* __restrict__ wkb, unsigned short* __restrict__ wvb,
                             unsigned short* __restrict__ w1b, unsigned short* __restrict__ w2b) {
  int i = blockIdx.x * 256 + threadIdx.x;
  if (i < 65536)                wkb[i] = f2bf(ca_w_in[65536 + i]);          // wk = ca_w_in[256:512]
  else if (i < 131072)          wvb[i - 65536] = f2bf(ca_w_in[131072 + (i - 65536)]); // wv
  else if (i < 131072 + 524288) w1b[i - 131072] = f2bf(w1[i - 131072]);
  else                          w2b[i - 655360] = f2bf(w2[i - 655360]);
}

// ---------------- kernel 2: SA QKV projection (plain f32, tiny) ----------------
__global__ void sa_qkv_kernel(const float* __restrict__ tgt, const float* __restrict__ qpos,
                              const float* __restrict__ w_in, const float* __restrict__ b_in,
                              float* __restrict__ qkv) {
  int o = blockIdx.x * 256 + threadIdx.x;           // 36*768 outputs
  int row = o / 768, c = o % 768;
  const float* w = w_in + (size_t)c * D;
  float s = b_in[c];
  bool useqp = (c < 512);                            // q,k use x+query_pos ; v uses x
  for (int k = 0; k < D; ++k) {
    float xv = tgt[row * D + k] + (useqp ? qpos[row * D + k] : 0.f);
    s += xv * w[k];
  }
  qkv[row * 768 + c] = s;
}

// ---------------- kernel 3: SA attention per head ----------------
__global__ void sa_attn_kernel(const float* __restrict__ qkv, float* __restrict__ satt) {
  __shared__ float sc[NQ][NQ];
  int h = blockIdx.x, tid = threadIdx.x;
  for (int idx = tid; idx < NQ * NQ; idx += 128) {
    int i = idx / NQ, j = idx % NQ;
    float s = 0.f;
    for (int k = 0; k < DH; ++k)
      s += qkv[i * 768 + h * DH + k] * qkv[j * 768 + 256 + h * DH + k];
    sc[i][j] = s * 0.17677669529663687f;             // 1/sqrt(32)
  }
  __syncthreads();
  if (tid < NQ) {
    float m = -3e30f;
    for (int j = 0; j < NQ; ++j) m = fmaxf(m, sc[tid][j]);
    float l = 0.f;
    for (int j = 0; j < NQ; ++j) { float p = __expf(sc[tid][j] - m); sc[tid][j] = p; l += p; }
    float rl = 1.f / l;
    for (int j = 0; j < NQ; ++j) sc[tid][j] *= rl;
  }
  __syncthreads();
  for (int idx = tid; idx < NQ * DH; idx += 128) {
    int i = idx / DH, d = idx % DH;
    float o = 0.f;
    for (int j = 0; j < NQ; ++j) o += sc[i][j] * qkv[j * 768 + 512 + h * DH + d];
    satt[i * D + h * DH + d] = o;
  }
}

// ---------------- kernel 4: SA out-proj + LN1 + CA Q-projection (scaled, bf16) --------
__global__ void sa_post_kernel(const float* __restrict__ tgt, const float* __restrict__ qpos,
                               const float* __restrict__ satt,
                               const float* __restrict__ w_out, const float* __restrict__ b_out,
                               const float* __restrict__ g1, const float* __restrict__ bt1,
                               const float* __restrict__ ca_w_in, const float* __restrict__ ca_b_in,
                               float* __restrict__ x1, unsigned short* __restrict__ qca) {
  __shared__ float co[NQ][D];
  __shared__ float mr[NQ], vr[NQ];
  int tid = threadIdx.x;
  for (int idx = tid; idx < NQ * D; idx += 256) {
    int row = idx >> 8, d = idx & 255;
    float s = b_out[d];
    for (int k = 0; k < D; ++k) s += satt[row * D + k] * w_out[d * D + k];
    co[row][d] = tgt[row * D + d] + s;               // residual
  }
  __syncthreads();
  if (tid < NQ) {
    float m = 0.f;
    for (int d = 0; d < D; ++d) m += co[tid][d];
    m *= (1.f / D);
    float v = 0.f;
    for (int d = 0; d < D; ++d) { float t = co[tid][d] - m; v += t * t; }
    mr[tid] = m; vr[tid] = v * (1.f / D);
  }
  __syncthreads();
  for (int idx = tid; idx < NQ * D; idx += 256) {
    int row = idx >> 8, d = idx & 255;
    float y = (co[row][d] - mr[row]) * rsqrtf(vr[row] + 1e-5f) * g1[d] + bt1[d];
    x1[row * D + d] = y;
  }
  __syncthreads();
  for (int idx = tid; idx < NQ * D; idx += 256) {   // q2 = x1 + query_pos (stage in co)
    int row = idx >> 8, d = idx & 255;
    co[row][d] = x1[row * D + d] + qpos[row * D + d];
  }
  __syncthreads();
  for (int idx = tid; idx < NQ * D; idx += 256) {   // CA wq projection, scaled by 1/sqrt(dh)
    int row = idx >> 8, c = idx & 255;
    float s = ca_b_in[c];
    for (int k = 0; k < D; ++k) s += co[row][k] * ca_w_in[c * D + k];
    qca[row * D + c] = f2bf(s * 0.17677669529663687f);
  }
  for (int idx = tid; idx < (QPAD - NQ) * D; idx += 256)  // zero pad rows 36..47
    qca[(NQ + idx / D) * D + (idx % D)] = 0;
}

// ---------------- kernel 5: K/V projection, WMMA bf16 (K row-major, V transposed) -----
__global__ void kv_kernel(const float* __restrict__ memv, const float* __restrict__ pos,
                          const unsigned short* __restrict__ wkb,
                          const unsigned short* __restrict__ wvb,
                          const float* __restrict__ ca_b_in,
                          unsigned short* __restrict__ Kb, unsigned short* __restrict__ Vt) {
  __shared__ unsigned short xk[32 * D];   // bf16(mem+pos)
  __shared__ unsigned short xv[32 * D];   // bf16(mem)
  int tid = threadIdx.x;
  int wave = tid >> 5, lane = tid & 31, lrow = lane & 15, lhalf = lane >> 4;
  size_t rowbase = (size_t)blockIdx.x * 32;
  for (int idx = tid; idx < 32 * D; idx += 256) {
    int r = idx / D, c = idx % D;
    size_t g = (rowbase + r) * D + c;
    float m = memv[g], p = pos[g];
    xk[idx] = f2bf(m + p);
    xv[idx] = f2bf(m);
  }
  __syncthreads();
#pragma unroll 1
  for (int i = 0; i < 8; ++i) {
    int t = i * 8 + wave;                 // 64 tiles: 2 mats x 2 mtiles x 16 ntiles
    int mat = t >> 5, rem = t & 31, mtile = rem >> 4, ntile = rem & 15;
    const unsigned short* X = mat ? xv : xk;
    const unsigned short* W = mat ? wvb : wkb;
    int n = ntile * 16 + lrow;
    v8f acc = vzero8();
#pragma unroll
    for (int kb = 0; kb < 8; ++kb) {
      int k0 = kb * 32;
      v16bf a = load_a16(X, D, mtile * 16 + lrow, k0, lhalf);
      v16bf b = *(const v16bf*)(W + (size_t)n * D + k0 + 16 * lhalf);
      acc = wmma_bf16(a, b, acc);
    }
    float bias = mat ? ca_b_in[512 + n] : ca_b_in[256 + n];
#pragma unroll
    for (int r = 0; r < 8; ++r) {
      size_t rg = rowbase + mtile * 16 + r + 8 * lhalf;
      float v = acc[r] + bias;
      if (mat == 0) Kb[rg * D + n] = f2bf(v);
      else          Vt[(size_t)n * STOT + rg] = f2bf(v);
    }
  }
}

// ---------------- kernel 6: flash cross-attention partials over shared keys ----------
__global__ void ca_kernel(const unsigned short* __restrict__ qca,
                          const unsigned short* __restrict__ Kb,
                          const unsigned short* __restrict__ Vt,
                          const unsigned char* __restrict__ kpm,
                          float* __restrict__ pO, float* __restrict__ pm, float* __restrict__ pl) {
  __shared__ float sO[8][3][16][32];       // per-wave partial O (first 8KB aliased as P stage)
  __shared__ float sM[8][3][16];
  __shared__ float sL[8][3][16];
  __shared__ float mtot[QPAD], ltot[QPAD];
  int tid = threadIdx.x;
  int wave = tid >> 5, lane = tid & 31, lrow = lane & 15, lhalf = lane >> 4;
  int chunk = blockIdx.x, h = blockIdx.y;
  unsigned short* pw = ((unsigned short*)&sO[0][0][0][0]) + wave * 512;  // 16x32 bf16/wave

  v16bf Aq[3];
#pragma unroll
  for (int mt = 0; mt < 3; ++mt)
    Aq[mt] = load_a16(qca, D, mt * 16 + lrow, h * DH, lhalf);

  float m_run[3][8], l_run[3][8];
  v8f o[3][2];
#pragma unroll
  for (int mt = 0; mt < 3; ++mt) {
#pragma unroll
    for (int r = 0; r < 8; ++r) { m_run[mt][r] = -1e30f; l_run[mt][r] = 0.f; }
    o[mt][0] = vzero8(); o[mt][1] = vzero8();
  }

#pragma unroll 1
  for (int g = 0; g < 4; ++g) {
    int kb = chunk * CHUNK + wave * 128 + g * 32;
    if (g < 3) __builtin_prefetch(Kb + (size_t)(kb + 32 + lrow) * D + h * DH, 0, 1);
    v16bf b0 = *(const v16bf*)(Kb + (size_t)(kb + lrow) * D + h * DH + 16 * lhalf);
    v16bf b1 = *(const v16bf*)(Kb + (size_t)(kb + 16 + lrow) * D + h * DH + 16 * lhalf);
    bool mk0 = kpm[kb + lrow] != 0;
    bool mk1 = kpm[kb + 16 + lrow] != 0;
#pragma unroll 1
    for (int mt = 0; mt < 3; ++mt) {
      v8f z = vzero8();
      v8f s0 = wmma_bf16(Aq[mt], b0, z);
      v8f s1 = wmma_bf16(Aq[mt], b1, z);
#pragma unroll
      for (int r = 0; r < 8; ++r) {
        if (mk0) s0[r] = -1e9f;
        if (mk1) s1[r] = -1e9f;
      }
#pragma unroll
      for (int r = 0; r < 8; ++r) {
        float mx = fmaxf(s0[r], s1[r]);
        mx = fmaxf(mx, __shfl_xor(mx, 1, 32));
        mx = fmaxf(mx, __shfl_xor(mx, 2, 32));
        mx = fmaxf(mx, __shfl_xor(mx, 4, 32));
        mx = fmaxf(mx, __shfl_xor(mx, 8, 32));
        float mn = fmaxf(m_run[mt][r], mx);
        float scl = __expf(m_run[mt][r] - mn);
        float p0 = __expf(s0[r] - mn);
        float p1 = __expf(s1[r] - mn);
        float rs = p0 + p1;
        rs += __shfl_xor(rs, 1, 32);
        rs += __shfl_xor(rs, 2, 32);
        rs += __shfl_xor(rs, 4, 32);
        rs += __shfl_xor(rs, 8, 32);
        l_run[mt][r] = l_run[mt][r] * scl + rs;
        m_run[mt][r] = mn;
        o[mt][0][r] *= scl;
        o[mt][1][r] *= scl;
        pw[(r + 8 * lhalf) * 32 + lrow] = f2bf(p0);
        pw[(r + 8 * lhalf) * 32 + 16 + lrow] = f2bf(p1);
      }
      __syncthreads();                                 // P staged in LDS
      v16bf Ap = load_a16(pw, 32, lrow, 0, lhalf);
#pragma unroll
      for (int n = 0; n < 2; ++n) {
        v16bf bv = *(const v16bf*)(Vt + (size_t)(h * DH + n * 16 + lrow) * STOT + kb + 16 * lhalf);
        o[mt][n] = wmma_bf16(Ap, bv, o[mt][n]);
      }
      __syncthreads();                                 // before P stage is rewritten
    }
  }
  __syncthreads();                                     // release pbuf alias of sO
#pragma unroll
  for (int mt = 0; mt < 3; ++mt) {
#pragma unroll
    for (int n = 0; n < 2; ++n)
#pragma unroll
      for (int r = 0; r < 8; ++r)
        sO[wave][mt][r + 8 * lhalf][n * 16 + lrow] = o[mt][n][r];
    if (lrow == 0)
#pragma unroll
      for (int r = 0; r < 8; ++r) {
        sM[wave][mt][r + 8 * lhalf] = m_run[mt][r];
        sL[wave][mt][r + 8 * lhalf] = l_run[mt][r];
      }
  }
  __syncthreads();
  if (tid < QPAD) {
    int mt = tid / 16, rr = tid % 16;
    float m = -3e30f;
    for (int w = 0; w < 8; ++w) m = fmaxf(m, sM[w][mt][rr]);
    float l = 0.f;
    for (int w = 0; w < 8; ++w) l += sL[w][mt][rr] * __expf(sM[w][mt][rr] - m);
    mtot[tid] = m; ltot[tid] = l;
    pm[(h * NCHUNK + chunk) * QPAD + tid] = m;
    pl[(h * NCHUNK + chunk) * QPAD + tid] = l;
  }
  __syncthreads();
  for (int idx = tid; idx < QPAD * 32; idx += 256) {
    int row = idx >> 5, col = idx & 31;
    int mt = row / 16, rr = row % 16;
    float acc = 0.f;
    for (int w = 0; w < 8; ++w)
      acc += sO[w][mt][rr][col] * __expf(sM[w][mt][rr] - mtot[row]);
    pO[((size_t)(h * NCHUNK + chunk) * QPAD + row) * 32 + col] = acc;
  }
}

// ---------------- kernel 7: merge chunk partials + per-expert window keys ------------
__global__ void comb_kernel(const unsigned short* __restrict__ qca,
                            const unsigned short* __restrict__ Kb,
                            const unsigned short* __restrict__ Vt,
                            const unsigned char* __restrict__ kpm,
                            const float* __restrict__ pO, const float* __restrict__ pm,
                            const float* __restrict__ pl, float* __restrict__ attn) {
  __shared__ float Os[QPAD][32];
  __shared__ float ms[QPAD], ls[QPAD];
  __shared__ float sw[NQ][WORD];
  int tid = threadIdx.x;
  int e = blockIdx.x, h = blockIdx.y;
  if (tid < QPAD) {
    float m = -3e30f;
    for (int c = 0; c < NCHUNK; ++c) m = fmaxf(m, pm[(h * NCHUNK + c) * QPAD + tid]);
    float l = 0.f;
    for (int c = 0; c < NCHUNK; ++c)
      l += pl[(h * NCHUNK + c) * QPAD + tid] * __expf(pm[(h * NCHUNK + c) * QPAD + tid] - m);
    ms[tid] = m; ls[tid] = l;
  }
  __syncthreads();
  for (int idx = tid; idx < QPAD * 32; idx += 256) {
    int row = idx >> 5, col = idx & 31;
    float acc = 0.f;
    for (int c = 0; c < NCHUNK; ++c)
      acc += pO[((size_t)(h * NCHUNK + c) * QPAD + row) * 32 + col] *
             __expf(pm[(h * NCHUNK + c) * QPAD + row] - ms[row]);
    Os[row][col] = acc;
  }
  __syncthreads();
  for (int idx = tid; idx < NQ * WORD; idx += 256) {
    int row = idx / WORD, j = idx % WORD;
    int key = SSUB + e * WORD + j;
    float s = 0.f;
    for (int k = 0; k < DH; ++k)
      s += bf2f(qca[row * D + h * DH + k]) * bf2f(Kb[(size_t)key * D + h * DH + k]);
    if (kpm[key]) s = -1e9f;
    sw[row][j] = s;
  }
  __syncthreads();
  for (int idx = tid; idx < NQ * 32; idx += 256) {
    int row = idx >> 5, col = idx & 31;
    float mw = -3e30f;
    for (int j = 0; j < WORD; ++j) mw = fmaxf(mw, sw[row][j]);
    float mf = fmaxf(ms[row], mw);
    float l = ls[row] * __expf(ms[row] - mf);
    float acc = Os[row][col] * __expf(ms[row] - mf);
    for (int j = 0; j < WORD; ++j) {
      float p = __expf(sw[row][j] - mf);
      l += p;
      acc += p * bf2f(Vt[(size_t)(h * DH + col) * STOT + SSUB + e * WORD + j]);
    }
    attn[((size_t)e * NQ + row) * D + h * DH + col] = acc / l;
  }
}

// ---------------- kernel 8: CA out-proj + residual + LN2 ----------------
__global__ void post2_kernel(const float* __restrict__ attn, const float* __restrict__ x1,
                             const float* __restrict__ w_out, const float* __restrict__ b_out,
                             const float* __restrict__ g2, const float* __restrict__ bt2,
                             float* __restrict__ x2f, unsigned short* __restrict__ x2b) {
  __shared__ float co[NQ][D];
  __shared__ float mr[NQ], vr[NQ];
  int e = blockIdx.x, tid = threadIdx.x;
  for (int idx = tid; idx < NQ * D; idx += 256) {
    int row = idx >> 8, d = idx & 255;
    float s = b_out[d];
    const float* ar = attn + ((size_t)e * NQ + row) * D;
    for (int k = 0; k < D; ++k) s += ar[k] * w_out[d * D + k];
    co[row][d] = x1[row * D + d] + s;
  }
  __syncthreads();
  if (tid < NQ) {
    float m = 0.f;
    for (int d = 0; d < D; ++d) m += co[tid][d];
    m *= (1.f / D);
    float v = 0.f;
    for (int d = 0; d < D; ++d) { float t = co[tid][d] - m; v += t * t; }
    mr[tid] = m; vr[tid] = v * (1.f / D);
  }
  __syncthreads();
  for (int idx = tid; idx < NQ * D; idx += 256) {
    int row = idx >> 8, d = idx & 255;
    float y = (co[row][d] - mr[row]) * rsqrtf(vr[row] + 1e-5f) * g2[d] + bt2[d];
    x2f[((size_t)e * NQ + row) * D + d] = y;
    x2b[((size_t)e * QPAD + row) * D + d] = f2bf(y);
  }
  for (int idx = tid; idx < (QPAD - NQ) * D; idx += 256)
    x2b[((size_t)e * QPAD + NQ + idx / D) * D + (idx % D)] = 0;
}

// ---------------- kernel 9: FFN layer 1 (WMMA, relu) ----------------
__global__ void ffn1_kernel(const unsigned short* __restrict__ x2b,
                            const unsigned short* __restrict__ w1b,
                            const float* __restrict__ b1, unsigned short* __restrict__ hid) {
  int tid = threadIdx.x;
  int wave = tid >> 5, lane = tid & 31, lrow = lane & 15, lhalf = lane >> 4;
  int nb = blockIdx.x, e = blockIdx.y;
  const unsigned short* X = x2b + (size_t)e * QPAD * D;
#pragma unroll 1
  for (int i = 0; i < 6; ++i) {
    int t = i * 8 + wave;                   // 48 tiles: 3 mt x 16 nt
    int mt = t >> 4, nt = t & 15;
    int n = nb * 256 + nt * 16 + lrow;
    v8f acc = vzero8();
#pragma unroll
    for (int kb = 0; kb < 8; ++kb) {
      int k0 = kb * 32;
      v16bf a = load_a16(X, D, mt * 16 + lrow, k0, lhalf);
      v16bf b = *(const v16bf*)(w1b + (size_t)n * D + k0 + 16 * lhalf);
      acc = wmma_bf16(a, b, acc);
    }
    float bias = b1[n];
#pragma unroll
    for (int r = 0; r < 8; ++r) {
      float v = acc[r] + bias;
      v = v > 0.f ? v : 0.f;
      hid[((size_t)e * QPAD + mt * 16 + r + 8 * lhalf) * DFF + n] = f2bf(v);
    }
  }
}

// ---------------- kernel 10: FFN layer 2 (WMMA) + residual + LN3 + transpose store ----
__global__ void ffn2_kernel(const unsigned short* __restrict__ hid,
                            const unsigned short* __restrict__ w2b,
                            const float* __restrict__ b2, const float* __restrict__ x2f,
                            const float* __restrict__ g3, const float* __restrict__ bt3,
                            float* __restrict__ out) {
  __shared__ float ff[QPAD][D];
  __shared__ float mr[NQ], vr[NQ];
  int tid = threadIdx.x;
  int wave = tid >> 5, lane = tid & 31, lrow = lane & 15, lhalf = lane >> 4;
  int e = blockIdx.x;
  const unsigned short* Xh = hid + (size_t)e * QPAD * DFF;
#pragma unroll 1
  for (int i = 0; i < 6; ++i) {
    int t = i * 8 + wave;
    int mt = t >> 4, nt = t & 15;
    int n = nt * 16 + lrow;
    v8f acc = vzero8();
#pragma unroll 1
    for (int kb = 0; kb < 64; ++kb) {
      int k0 = kb * 32;
      __builtin_prefetch(w2b + (size_t)n * DFF + k0 + 64, 0, 1);
      v16bf a = load_a16(Xh, DFF, mt * 16 + lrow, k0, lhalf);
      v16bf b = *(const v16bf*)(w2b + (size_t)n * DFF + k0 + 16 * lhalf);
      acc = wmma_bf16(a, b, acc);
    }
    float bias = b2[n];
#pragma unroll
    for (int r = 0; r < 8; ++r)
      ff[mt * 16 + r + 8 * lhalf][n] = acc[r] + bias;
  }
  __syncthreads();
  if (tid < NQ) {
    float sum = 0.f, sq = 0.f;
    for (int d = 0; d < D; ++d) {
      float v = x2f[((size_t)e * NQ + tid) * D + d] + ff[tid][d];
      sum += v; sq += v * v;
    }
    float m = sum * (1.f / D);
    mr[tid] = m; vr[tid] = sq * (1.f / D) - m * m;
  }
  __syncthreads();
  for (int idx = tid; idx < NQ * D; idx += 256) {
    int row = idx >> 8, d = idx & 255;
    float v = x2f[((size_t)e * NQ + row) * D + d] + ff[row][d];
    float y = (v - mr[row]) * rsqrtf(vr[row] + 1e-5f) * g3[d] + bt3[d];
    out[((size_t)row * NE + e) * D + d] = y;       // output (q, e, d)
  }
}

// ---------------- host-side launch ----------------
extern "C" void kernel_launch(void* const* d_in, const int* in_sizes, int n_in,
                              void* d_out, int out_size, void* d_ws, size_t ws_size,
                              hipStream_t stream) {
  (void)in_sizes; (void)n_in; (void)out_size; (void)ws_size;
  const float* tgt      = (const float*)d_in[0];
  const float* memv     = (const float*)d_in[1];
  const unsigned char* kpm = (const unsigned char*)d_in[2];
  const float* pos      = (const float*)d_in[3];
  const float* qpos     = (const float*)d_in[4];
  const float* sa_w_in  = (const float*)d_in[6];
  const float* sa_b_in  = (const float*)d_in[7];
  const float* sa_w_out = (const float*)d_in[8];
  const float* sa_b_out = (const float*)d_in[9];
  const float* ca_w_in  = (const float*)d_in[10];
  const float* ca_b_in  = (const float*)d_in[11];
  const float* ca_w_out = (const float*)d_in[12];
  const float* ca_b_out = (const float*)d_in[13];
  const float* w1       = (const float*)d_in[14];
  const float* b1       = (const float*)d_in[15];
  const float* w2       = (const float*)d_in[16];
  const float* b2       = (const float*)d_in[17];
  const float* ln1_g = (const float*)d_in[18];
  const float* ln1_b = (const float*)d_in[19];
  const float* ln2_g = (const float*)d_in[20];
  const float* ln2_b = (const float*)d_in[21];
  const float* ln3_g = (const float*)d_in[22];
  const float* ln3_b = (const float*)d_in[23];

  char* ws = (char*)d_ws;
  size_t off = 0;
  auto take = [&](size_t bytes) -> void* {
    void* p = ws + off;
    off += (bytes + 255) & ~(size_t)255;
    return p;
  };
  float*          qkv  = (float*)take((size_t)NQ * 768 * 4);
  float*          satt = (float*)take((size_t)NQ * D * 4);
  float*          x1   = (float*)take((size_t)NQ * D * 4);
  unsigned short* qca  = (unsigned short*)take((size_t)QPAD * D * 2);
  unsigned short* wkb  = (unsigned short*)take((size_t)D * D * 2);
  unsigned short* wvb  = (unsigned short*)take((size_t)D * D * 2);
  unsigned short* w1b  = (unsigned short*)take((size_t)DFF * D * 2);
  unsigned short* w2b  = (unsigned short*)take((size_t)D * DFF * 2);
  unsigned short* Kb   = (unsigned short*)take((size_t)STOT * D * 2);
  unsigned short* Vt   = (unsigned short*)take((size_t)STOT * D * 2);
  float*          pO   = (float*)take((size_t)H * NCHUNK * QPAD * 32 * 4);
  float*          pm   = (float*)take((size_t)H * NCHUNK * QPAD * 4);
  float*          pl   = (float*)take((size_t)H * NCHUNK * QPAD * 4);
  float*          attn = (float*)take((size_t)NE * NQ * D * 4);
  float*          x2f  = (float*)take((size_t)NE * NQ * D * 4);
  unsigned short* x2b  = (unsigned short*)take((size_t)NE * QPAD * D * 2);
  unsigned short* hid  = (unsigned short*)take((size_t)NE * QPAD * DFF * 2);
  float* out = (float*)d_out;

  wconv_kernel<<<4608, 256, 0, stream>>>(ca_w_in, w1, w2, wkb, wvb, w1b, w2b);
  sa_qkv_kernel<<<108, 256, 0, stream>>>(tgt, qpos, sa_w_in, sa_b_in, qkv);
  sa_attn_kernel<<<H, 128, 0, stream>>>(qkv, satt);
  sa_post_kernel<<<1, 256, 0, stream>>>(tgt, qpos, satt, sa_w_out, sa_b_out,
                                        ln1_g, ln1_b, ca_w_in, ca_b_in, x1, qca);
  kv_kernel<<<STOT / 32, 256, 0, stream>>>(memv, pos, wkb, wvb, ca_b_in, Kb, Vt);
  ca_kernel<<<dim3(NCHUNK, H), 256, 0, stream>>>(qca, Kb, Vt, kpm, pO, pm, pl);
  comb_kernel<<<dim3(NE, H), 256, 0, stream>>>(qca, Kb, Vt, kpm, pO, pm, pl, attn);
  post2_kernel<<<NE, 256, 0, stream>>>(attn, x1, ca_w_out, ca_b_out, ln2_g, ln2_b, x2f, x2b);
  ffn1_kernel<<<dim3(DFF / 256, NE), 256, 0, stream>>>(x2b, w1b, b1, hid);
  ffn2_kernel<<<NE, 256, 0, stream>>>(hid, w2b, b2, x2f, ln3_g, ln3_b, out);
}
